// CausalSelfAttention_36395552866808
// MI455X (gfx1250) — matrix operand order repaired and checked by reading, not compiled, not run on hardware
//
#include <hip/hip_runtime.h>
#include <math.h>

typedef __bf16 bf16;
typedef __attribute__((ext_vector_type(16))) __bf16 v16bf;
typedef __attribute__((ext_vector_type(8)))  __bf16 v8bf;
typedef __attribute__((ext_vector_type(8)))  float  v8f;

#define BB   4
#define TT   2048
#define CC   1024
#define HH   16
#define HD   64
#define NTOK (BB*TT)     /* 8192 */
#define N3C  (3*CC)      /* 3072 */

// ---------------------------------------------------------------------------
// CDNA5 async / TDM availability (guarded: falls back to sync staging)
// ---------------------------------------------------------------------------
#if __has_builtin(__builtin_amdgcn_global_load_async_to_lds_b128)
#define HAS_ASYNC 1
#else
#define HAS_ASYNC 0
#endif

#if __has_builtin(__builtin_amdgcn_tensor_load_to_lds) && \
    __has_builtin(__builtin_amdgcn_s_wait_tensorcnt)
#define HAS_TDM 1
#else
#define HAS_TDM 0
#endif

// The async builtin takes pointers to int32x4 in AS(1)/AS(3).
typedef int v4i __attribute__((ext_vector_type(4)));
typedef __attribute__((address_space(1))) v4i g1_v4i;
typedef __attribute__((address_space(3))) v4i l3_v4i;

static __device__ __forceinline__ g1_v4i* to_g1(const void* p) {
  return (g1_v4i*)(uintptr_t)p;           // int->ptr, AS1
}
static __device__ __forceinline__ l3_v4i* to_l3(void* p) {
  // generic LDS address: low 32 bits are the LDS byte offset (ISA 10.2)
  return (l3_v4i*)(unsigned)(uintptr_t)p;
}

#if HAS_ASYNC
#if __has_builtin(__builtin_amdgcn_s_wait_asynccnt)
#define WAIT_ASYNC(n) __builtin_amdgcn_s_wait_asynccnt(n)
#else
#define WAIT_ASYNC(n) asm volatile("s_wait_asynccnt %0" ::"i"(n) : "memory")
#endif
#else
#define WAIT_ASYNC(n) do { } while (0)
#endif

// Stage 16 bf16 (32B) from global to LDS: async DMA if available.
static __device__ __forceinline__ void stage_row16(const bf16* g, bf16* l) {
#if HAS_ASYNC
  __builtin_amdgcn_global_load_async_to_lds_b128(to_g1(g),     to_l3(l),     0, 0);
  __builtin_amdgcn_global_load_async_to_lds_b128(to_g1(g + 8), to_l3(l + 8), 0, 0);
#else
  *(v8bf*)l       = *(const v8bf*)g;
  *(v8bf*)(l + 8) = *(const v8bf*)(g + 8);
#endif
}

#if HAS_TDM
typedef unsigned int u32x4 __attribute__((ext_vector_type(4)));
typedef int          i32x4 __attribute__((ext_vector_type(4)));
typedef int          i32x8 __attribute__((ext_vector_type(8)));

// 2D TDM tile load, D# packing per CDNA5 ISA.
// data_size = 2 bytes (bf16).  Issued once per wave; tracked by TENSORcnt.
static __device__ __forceinline__ void tdm_load_2d(uint64_t gaddr, unsigned lds,
                                                   unsigned td0, unsigned td1,
                                                   unsigned t0, unsigned t1,
                                                   unsigned stride0) {
  u32x4 g0;
  g0[0] = 1u;                                            // count=1 (valid), user mode
  g0[1] = lds;                                           // lds_addr
  g0[2] = (unsigned)gaddr;                               // global_addr[31:0]
  g0[3] = (unsigned)((gaddr >> 32) & 0x1FFFFFFu) | (2u << 30); // addr[56:32] | type=2
  i32x8 g1;
  g1[0] = (int)(1u << 16);                               // data_size=1 -> 2 bytes
  g1[1] = (int)((td0 & 0xFFFFu) << 16);                  // tensor_dim0[15:0]
  g1[2] = (int)((td0 >> 16) | ((td1 & 0xFFFFu) << 16));  // td0[31:16] | td1[15:0]
  g1[3] = (int)((td1 >> 16) | (t0 << 16));               // td1[31:16] | tile_dim0
  g1[4] = (int)t1;                                       // tile_dim1 | tile_dim2=0
  g1[5] = (int)stride0;                                  // tensor_dim0_stride[31:0]
  g1[6] = 0;                                             // stride0[47:32] | stride1 lo
  g1[7] = 0;
  i32x4 z4 = {};
#if __clang_major__ >= 23
  i32x8 z8 = {};
  __builtin_amdgcn_tensor_load_to_lds(g0, g1, z4, z4, z8, 0);
#else
  __builtin_amdgcn_tensor_load_to_lds(g0, g1, z4, z4, 0);
#endif
}
#endif // HAS_TDM

// ---------------------------------------------------------------------------
// WMMA helpers (layouts per CDNA5 ISA §7.12.2)
// ---------------------------------------------------------------------------
static __device__ __forceinline__ v8f wmma_bf16(v16bf a, v16bf b, v8f c) {
  return __builtin_amdgcn_wmma_f32_16x16x32_bf16(false, a, false, b, (short)0, c,
                                                 false, false);
}

// A-fragment (16x32 bf16): lane L holds row (L&15); K contiguous h*16..h*16+15
static __device__ __forceinline__ v16bf afrag_from_row(const bf16* rowp, int h) {
  v8bf lo = *(const v8bf*)(rowp + h * 16);
  v8bf hi = *(const v8bf*)(rowp + h * 16 + 8);
  return __builtin_shufflevector(lo, hi, 0,1,2,3,4,5,6,7,8,9,10,11,12,13,14,15);
}

// B-fragment (32x16 bf16) from [N][K]-major storage.
static __device__ __forceinline__ v16bf bfrag_nk(const bf16* base, int ld,
                                                 int ncol, int k0) {
  int lane = threadIdx.x & 31;
  int n = lane & 15, h = lane >> 4;
  return *(const v16bf*)(base + (size_t)(ncol + n) * ld + k0 + h * 16);
}

// ---------------------------------------------------------------------------
// Stage 0: casts
// ---------------------------------------------------------------------------
__global__ void cast_f32_to_bf16_v4(const float4* __restrict__ in,
                                    bf16* __restrict__ out, int n4) {
  int i = blockIdx.x * 256 + threadIdx.x;
  if (i < n4) {
    float4 f = in[i];
    int o = i * 4;
    out[o + 0] = (bf16)f.x; out[o + 1] = (bf16)f.y;
    out[o + 2] = (bf16)f.z; out[o + 3] = (bf16)f.w;
  }
}

__global__ void transpose_cast_bf16(const float* __restrict__ in,
                                    bf16* __restrict__ out, int rows, int cols) {
  int i = blockIdx.x * 256 + threadIdx.x;
  if (i < rows * cols) {
    int n = i / rows, k = i % rows;
    out[i] = (bf16)in[(size_t)k * cols + n];
  }
}

// ---------------------------------------------------------------------------
// Stage 1: QKV GEMM.  Double-buffered LDS, async tile DMA, bf16 WMMA.
// Block 128x128xK32; 8 waves (4 M x 2 N), wave tile 32x64.
// ---------------------------------------------------------------------------
__global__ __launch_bounds__(256)
void qkv_gemm(const bf16* __restrict__ A, const bf16* __restrict__ Bt,
              const float* __restrict__ bias,
              bf16* __restrict__ Q, bf16* __restrict__ K, bf16* __restrict__ VT) {
  __shared__ __attribute__((aligned(64))) bf16 Alds[2][128 * 32];
  __shared__ __attribute__((aligned(64))) bf16 Blds[2][128 * 32];

  const int bm = blockIdx.y * 128;
  const int bn = blockIdx.x * 128;
  const int tid  = threadIdx.x;
  const int wave = tid >> 5;
  const int lane = tid & 31;
  const int fl = lane & 15, fh = lane >> 4;
  const int wm = (wave & 3) * 32;
  const int wn = (wave >> 2) * 64;
  const int lr = tid >> 1;          // 0..127 tile row
  const int lc = (tid & 1) * 16;    // 0 or 16

  v8f acc[2][4] = {};

  const bf16* ga = A  + (size_t)(bm + lr) * CC + lc;
  const bf16* gb = Bt + (size_t)(bn + lr) * CC + lc;

  // prologue: stage tile 0 into buffer 0
  stage_row16(ga, &Alds[0][lr * 32 + lc]);
  stage_row16(gb, &Blds[0][lr * 32 + lc]);

  int buf = 0;
  for (int kb = 0; kb < CC; kb += 32) {
    if (kb + 32 < CC) {           // stage next tile into other buffer
      stage_row16(ga + kb + 32, &Alds[buf ^ 1][lr * 32 + lc]);
      stage_row16(gb + kb + 32, &Blds[buf ^ 1][lr * 32 + lc]);
      WAIT_ASYNC(4);              // current tile landed; next still in flight
    } else {
      WAIT_ASYNC(0);
    }
    __syncthreads();

    v16bf af[2];
#pragma unroll
    for (int i = 0; i < 2; ++i)
      af[i] = afrag_from_row(&Alds[buf][(wm + i * 16 + fl) * 32], fh);
#pragma unroll
    for (int j = 0; j < 4; ++j) {
      v16bf bf = bfrag_nk(Blds[buf], 32, wn + j * 16, 0);
#pragma unroll
      for (int i = 0; i < 2; ++i)
        acc[i][j] = wmma_bf16(af[i], bf, acc[i][j]);
    }
    __syncthreads();
    buf ^= 1;
  }

  // Epilogue: C-layout element e at (M = e + 8*fh, N = lane&15)
#pragma unroll
  for (int i = 0; i < 2; ++i)
#pragma unroll
    for (int j = 0; j < 4; ++j)
#pragma unroll
      for (int e = 0; e < 8; ++e) {
        int m = bm + wm + i * 16 + e + 8 * fh;
        int n = bn + wn + j * 16 + fl;
        float v = acc[i][j][e] + bias[n];
        int b = m >> 11, t = m & (TT - 1);
        int which = n >> 10;           // 0=q 1=k 2=v
        int c = n & (CC - 1);
        int hh = c >> 6, d = c & (HD - 1);
        size_t bh = (size_t)(b * HH + hh);
        if (which == 0)      Q[(bh * TT + t) * HD + d] = (bf16)(v * 0.125f);
        else if (which == 1) K[(bh * TT + t) * HD + d] = (bf16)v;
        else                 VT[(bh * HD + d) * TT + t] = (bf16)v;
      }
}

// ---------------------------------------------------------------------------
// Stage 2: flash attention.  4 waves x 16 q rows; 64-key K/V^T tiles in LDS
// staged by the Tensor Data Mover (TENSORcnt) when available.
// ---------------------------------------------------------------------------
__global__ __launch_bounds__(128)
void attn_flash(const bf16* __restrict__ Q, const bf16* __restrict__ K,
                const bf16* __restrict__ VT, bf16* __restrict__ Y) {
  __shared__ __attribute__((aligned(64))) bf16 Klds[64 * 64];
  __shared__ __attribute__((aligned(64))) bf16 Vlds[64 * 64];
  __shared__ __attribute__((aligned(64))) bf16 Plds[4 * 16 * 64];

  const int bh   = blockIdx.y;
  const int qb   = blockIdx.x * 64;
  const int tid  = threadIdx.x;
  const int wave = tid >> 5;
  const int lane = tid & 31;
  const int fl = lane & 15, fh = lane >> 4;

  const bf16* Qb = Q  + (size_t)bh * TT * HD;
  const bf16* Kb = K  + (size_t)bh * TT * HD;
  const bf16* Vb = VT + (size_t)bh * HD * TT;

  const bf16* qrow = Qb + (size_t)(qb + wave * 16 + fl) * HD;
  v16bf qf0 = afrag_from_row(qrow, fh);
  v16bf qf1 = afrag_from_row(qrow + 32, fh);

  v8f o[4] = {};
  float m_i[8], l_i[8];
#pragma unroll
  for (int e = 0; e < 8; ++e) { m_i[e] = -INFINITY; l_i[e] = 0.f; }

  const int lr = tid >> 1;          // 0..63
  const int lc = (tid & 1) * 32;    // 0 or 32
  (void)lr; (void)lc;

  for (int kb = 0; kb <= qb; kb += 64) {
#if HAS_TDM
    __syncthreads();                // readers of previous tile are done
    if (wave == 0) {
      // K tile: [64 keys][64 hd], row stride HD (contiguous block)
      tdm_load_2d((uint64_t)(uintptr_t)(Kb + (size_t)kb * HD),
                  (unsigned)(uintptr_t)&Klds[0], HD, TT, 64, 64, HD);
      // V^T tile: [64 hd][64 keys], row stride T (strided tile)
      tdm_load_2d((uint64_t)(uintptr_t)(Vb + kb),
                  (unsigned)(uintptr_t)&Vlds[0], TT, HD, 64, 64, TT);
      __builtin_amdgcn_s_wait_tensorcnt(0);
    }
    __syncthreads();
    if (kb + 64 <= qb) __builtin_prefetch(Kb + (size_t)(kb + 64) * HD, 0, 1);
#else
    const bf16* gk = Kb + (size_t)(kb + lr) * HD + lc;
    const bf16* gv = Vb + (size_t)lr * TT + kb + lc;
    v8bf k0 = *(const v8bf*)gk,        k1 = *(const v8bf*)(gk + 8);
    v8bf k2 = *(const v8bf*)(gk + 16), k3 = *(const v8bf*)(gk + 24);
    v8bf v0 = *(const v8bf*)gv,        v1 = *(const v8bf*)(gv + 8);
    v8bf v2 = *(const v8bf*)(gv + 16), v3 = *(const v8bf*)(gv + 24);
    if (kb + 64 <= qb) {
      __builtin_prefetch(gk + 64 * HD, 0, 1);
      __builtin_prefetch(gv + 64, 0, 1);
    }
    __syncthreads();
    *(v8bf*)&Klds[lr * 64 + lc]      = k0; *(v8bf*)&Klds[lr * 64 + lc + 8]  = k1;
    *(v8bf*)&Klds[lr * 64 + lc + 16] = k2; *(v8bf*)&Klds[lr * 64 + lc + 24] = k3;
    *(v8bf*)&Vlds[lr * 64 + lc]      = v0; *(v8bf*)&Vlds[lr * 64 + lc + 8]  = v1;
    *(v8bf*)&Vlds[lr * 64 + lc + 16] = v2; *(v8bf*)&Vlds[lr * 64 + lc + 24] = v3;
    __syncthreads();
#endif

    // S = Q K^T : K rows [key][hd] are [N][K]-major for the B operand
    v8f s[4];
#pragma unroll
    for (int j = 0; j < 4; ++j) {
      v8f z = {};
      z = wmma_bf16(qf0, bfrag_nk(Klds, 64, j * 16, 0),  z);
      z = wmma_bf16(qf1, bfrag_nk(Klds, 64, j * 16, 32), z);
      s[j] = z;
    }

    if (kb == qb) {   // diagonal tile: causal mask
#pragma unroll
      for (int j = 0; j < 4; ++j)
#pragma unroll
        for (int e = 0; e < 8; ++e)
          if (kb + j * 16 + fl > qb + wave * 16 + e + 8 * fh)
            s[j][e] = -INFINITY;
    }

    // online softmax (row = e + 8*fh spans one 16-lane half)
#pragma unroll
    for (int e = 0; e < 8; ++e) {
      float mx = s[0][e];
#pragma unroll
      for (int j = 1; j < 4; ++j) mx = fmaxf(mx, s[j][e]);
#pragma unroll
      for (int d = 1; d < 16; d <<= 1) mx = fmaxf(mx, __shfl_xor(mx, d, 32));
      float mnew = fmaxf(m_i[e], mx);
      float corr = (m_i[e] == -INFINITY) ? 0.f : __expf(m_i[e] - mnew);
      float rsum = 0.f;
#pragma unroll
      for (int j = 0; j < 4; ++j) {
        float p = __expf(s[j][e] - mnew);
        s[j][e] = p;
        rsum += p;
      }
#pragma unroll
      for (int d = 1; d < 16; d <<= 1) rsum += __shfl_xor(rsum, d, 32);
      l_i[e] = l_i[e] * corr + rsum;
      m_i[e] = mnew;
#pragma unroll
      for (int j = 0; j < 4; ++j) o[j][e] *= corr;
    }

    // P: C-layout -> LDS (wave-private, row-major [16][64]) -> A-fragments
    bf16* pw = &Plds[wave * 16 * 64];
#pragma unroll
    for (int j = 0; j < 4; ++j)
#pragma unroll
      for (int e = 0; e < 8; ++e)
        pw[(e + 8 * fh) * 64 + j * 16 + fl] = (bf16)s[j][e];

#pragma unroll
    for (int kc = 0; kc < 2; ++kc) {
      v16bf pf = afrag_from_row(pw + fl * 64 + kc * 32, fh);
#pragma unroll
      for (int j2 = 0; j2 < 4; ++j2)
        o[j2] = wmma_bf16(pf, bfrag_nk(Vlds, 64, j2 * 16, kc * 32), o[j2]);
    }
  }

  const int b = bh / HH, head = bh % HH;
#pragma unroll
  for (int e = 0; e < 8; ++e) {
    float inv = 1.f / l_i[e];
    int m = qb + wave * 16 + e + 8 * fh;
    bf16* yrow = Y + ((size_t)(b * TT + m)) * CC + head * HD;
#pragma unroll
    for (int j2 = 0; j2 < 4; ++j2)
      yrow[j2 * 16 + fl] = (bf16)(o[j2][e] * inv);
  }
}

// ---------------------------------------------------------------------------
// Stage 3: output projection (same pipeline as qkv_gemm, fp32 out + bias).
// ---------------------------------------------------------------------------
__global__ __launch_bounds__(256)
void proj_gemm(const bf16* __restrict__ A, const bf16* __restrict__ Bt,
               const float* __restrict__ bias, float* __restrict__ out) {
  __shared__ __attribute__((aligned(64))) bf16 Alds[2][128 * 32];
  __shared__ __attribute__((aligned(64))) bf16 Blds[2][128 * 32];

  const int bm = blockIdx.y * 128;
  const int bn = blockIdx.x * 128;
  const int tid  = threadIdx.x;
  const int wave = tid >> 5;
  const int lane = tid & 31;
  const int fl = lane & 15, fh = lane >> 4;
  const int wm = (wave & 3) * 32;
  const int wn = (wave >> 2) * 64;
  const int lr = tid >> 1;
  const int lc = (tid & 1) * 16;

  v8f acc[2][4] = {};

  const bf16* ga = A  + (size_t)(bm + lr) * CC + lc;
  const bf16* gb = Bt + (size_t)(bn + lr) * CC + lc;

  stage_row16(ga, &Alds[0][lr * 32 + lc]);
  stage_row16(gb, &Blds[0][lr * 32 + lc]);

  int buf = 0;
  for (int kb = 0; kb < CC; kb += 32) {
    if (kb + 32 < CC) {
      stage_row16(ga + kb + 32, &Alds[buf ^ 1][lr * 32 + lc]);
      stage_row16(gb + kb + 32, &Blds[buf ^ 1][lr * 32 + lc]);
      WAIT_ASYNC(4);
    } else {
      WAIT_ASYNC(0);
    }
    __syncthreads();

    v16bf af[2];
#pragma unroll
    for (int i = 0; i < 2; ++i)
      af[i] = afrag_from_row(&Alds[buf][(wm + i * 16 + fl) * 32], fh);
#pragma unroll
    for (int j = 0; j < 4; ++j) {
      v16bf bf = bfrag_nk(Blds[buf], 32, wn + j * 16, 0);
#pragma unroll
      for (int i = 0; i < 2; ++i)
        acc[i][j] = wmma_bf16(af[i], bf, acc[i][j]);
    }
    __syncthreads();
    buf ^= 1;
  }

#pragma unroll
  for (int i = 0; i < 2; ++i)
#pragma unroll
    for (int j = 0; j < 4; ++j)
#pragma unroll
      for (int e = 0; e < 8; ++e) {
        int m = bm + wm + i * 16 + e + 8 * fh;
        int n = bn + wn + j * 16 + fl;
        out[(size_t)m * CC + n] = acc[i][j][e] + bias[n];
      }
}

// ---------------------------------------------------------------------------
// Host launcher
// ---------------------------------------------------------------------------
extern "C" void kernel_launch(void* const* d_in, const int* in_sizes, int n_in,
                              void* d_out, int out_size, void* d_ws, size_t ws_size,
                              hipStream_t stream) {
  const float* x     = (const float*)d_in[0];
  const float* Wqkv  = (const float*)d_in[1];
  const float* bqkv  = (const float*)d_in[2];
  const float* Wproj = (const float*)d_in[3];
  const float* bproj = (const float*)d_in[4];
  float* out = (float*)d_out;

  char* ws = (char*)d_ws;
  size_t off = 0;
  bf16* xb     = (bf16*)(ws + off); off += (size_t)NTOK * CC * 2;
  bf16* wqkvT  = (bf16*)(ws + off); off += (size_t)N3C  * CC * 2;
  bf16* wprojT = (bf16*)(ws + off); off += (size_t)CC   * CC * 2;
  bf16* Qd     = (bf16*)(ws + off); off += (size_t)BB * HH * TT * HD * 2;
  bf16* Kd     = (bf16*)(ws + off); off += (size_t)BB * HH * TT * HD * 2;
  bf16* VTd    = (bf16*)(ws + off); off += (size_t)BB * HH * HD * TT * 2;
  bf16* Yd     = (bf16*)(ws + off); off += (size_t)NTOK * CC * 2;
  (void)ws_size; (void)in_sizes; (void)n_in; (void)out_size;

  {
    int n4 = NTOK * CC / 4;
    cast_f32_to_bf16_v4<<<n4 / 256, 256, 0, stream>>>((const float4*)x, xb, n4);
    int nq = N3C * CC;
    transpose_cast_bf16<<<(nq + 255) / 256, 256, 0, stream>>>(Wqkv, wqkvT, CC, N3C);
    int np = CC * CC;
    transpose_cast_bf16<<<(np + 255) / 256, 256, 0, stream>>>(Wproj, wprojT, CC, CC);
  }

  qkv_gemm<<<dim3(N3C / 128, NTOK / 128), 256, 0, stream>>>(xb, wqkvT, bqkv,
                                                            Qd, Kd, VTd);
  attn_flash<<<dim3(TT / 64, BB * HH), 128, 0, stream>>>(Qd, Kd, VTd, Yd);
  proj_gemm<<<dim3(CC / 128, NTOK / 128), 256, 0, stream>>>(Yd, wprojT, bproj, out);
}
